// KernelCustomFreezeConv_58188216926330
// MI455X (gfx1250) — matrix operand order repaired and checked by reading, not compiled, not run on hardware
//
#include <hip/hip_runtime.h>

// ---------------------------------------------------------------------------
// SincNet conv1d for MI455X (gfx1250, wave32, WMMA bf16 16x16x32).
// Strategy: error-compensated bf16 (hi/lo split, 3 WMMA products) implicit
// GEMM; all operands pre-packed into WMMA fragment layout in workspace and
// kept L2-resident (192 MB L2 >> 32 MB working set); no LDS in the hot loop.
// ---------------------------------------------------------------------------

typedef __attribute__((ext_vector_type(16))) __bf16 v16bf;
typedef __attribute__((ext_vector_type(8)))  float  v8f;

#define FS_      16000.0f
#define NFILT    80
#define FDIM     251
#define KPAD     256          // K padded to 8 * 32
#define LIN      32000
#define NBATCH   32
#define LOUT     (LIN - FDIM + 1)          // 31750
#define NT       256                        // output times per block
#define TTILES   ((LOUT + NT - 1) / NT)     // 125
#define MTILES   (NFILT / 16)               // 5
#define TWO_PI   6.283185307179586f

static __device__ __forceinline__ unsigned short f2bf(float f) {
  unsigned u = __float_as_uint(f);
  u += 0x7FFFu + ((u >> 16) & 1u);          // round-to-nearest-even
  return (unsigned short)(u >> 16);
}
static __device__ __forceinline__ float bf2f(unsigned short h) {
  return __uint_as_float(((unsigned)h) << 16);
}

// ---------------------------------------------------------------------------
// Phase 1a: build the 80x251 sinc filters (f32), zero-padded to 80x256.
// One block per filter; row-max reduction in LDS (jnp.max semantics).
// ---------------------------------------------------------------------------
__global__ void build_filters(const float* __restrict__ nf1,
                              const float* __restrict__ nf2,
                              const float* __restrict__ amp,
                              float* __restrict__ Fout) {
  int f = blockIdx.x;
  int p = threadIdx.x;                       // 0..255
  __shared__ float smax[256];

  float f1n = fabsf(nf1[f]) + 50.0f / FS_;
  float f2n = f1n + fabsf(nf2[f] - f1n) + 50.0f / FS_;
  float f1  = f1n * FS_;
  float f2  = f2n * FS_;
  float a   = fabsf(amp[f]);

  float bp = 0.0f;
  if (p < FDIM) {
    int d = p - (FDIM - 1) / 2;              // -125..125
    float s1 = 1.0f, s2 = 1.0f;
    if (d != 0) {
      float t  = (float)d / FS_;
      float a1 = TWO_PI * f1 * t;
      float a2 = TWO_PI * f2 * t;
      s1 = sinf(a1) / a1;                    // sinc is even -> sign-safe
      s2 = sinf(a2) / a2;
    }
    bp = a * (2.0f * f2 * s2 - 2.0f * f1 * s1);
  }

  smax[p] = (p < FDIM) ? bp : -3.4e38f;
  __syncthreads();
  for (int s = 128; s > 0; s >>= 1) {
    if (p < s) smax[p] = fmaxf(smax[p], smax[p + s]);
    __syncthreads();
  }
  float mx = smax[0];

  // window n = linspace(0, 251, 251) -> n_p = p*251/250 -> cos(2*pi*p/250)
  float w   = 0.54f - 0.46f * cosf(TWO_PI * (float)p / 250.0f);
  float val = (p < FDIM) ? (bp / mx) * w : 0.0f;
  Fout[f * KPAD + p] = val;
}

// ---------------------------------------------------------------------------
// Phase 1b: pre-swizzle filters into WMMA A-fragment layout (bf16 16x32),
// split into hi + lo(residual) planes.
// A layout (contiguous k, same as B): lane L -> M = L%16; VGPR v -> K pair at
//   (L/16)*16 + v*2  (elements k in [15:0], k+1 in [31:16]).
// ---------------------------------------------------------------------------
__global__ void build_afrag(const float* __restrict__ Fin,
                            unsigned* __restrict__ AH,
                            unsigned* __restrict__ AL) {
  int bx   = blockIdx.x;                     // 0..MTILES*8-1
  int mt   = bx >> 3;
  int kt   = bx & 7;
  int lane = threadIdx.x;                    // 32 threads
  int m    = mt * 16 + (lane & 15);
  int half = lane >> 4;
#pragma unroll
  for (int v = 0; v < 8; ++v) {
    int kb = half * 16 + v * 2;
    int k  = kt * 32 + kb;
    float x0 = Fin[m * KPAD + k];
    float x1 = Fin[m * KPAD + k + 1];
    unsigned short h0 = f2bf(x0), h1 = f2bf(x1);
    unsigned short l0 = f2bf(x0 - bf2f(h0));
    unsigned short l1 = f2bf(x1 - bf2f(h1));
    int idx = ((mt * 8 + kt) * 32 + lane) * 8 + v;
    AH[idx] = (unsigned)h0 | ((unsigned)h1 << 16);
    AL[idx] = (unsigned)l0 | ((unsigned)l1 << 16);
  }
}

// ---------------------------------------------------------------------------
// Phase 1c: pack x into "oct" arrays: Xoct[b*LIN+t] = 8 consecutive bf16
// (x[b,t..t+7], tail-clamped) in one uint4 -> every B-fragment half is one
// aligned 16B load. hi + lo residual planes. 32 MB total: L2-resident.
// ---------------------------------------------------------------------------
__global__ void build_xoct(const float* __restrict__ x,
                           uint4* __restrict__ XH,
                           uint4* __restrict__ XL) {
  int pos = blockIdx.x * blockDim.x + threadIdx.x;   // 0..NBATCH*LIN-1
  int b = pos / LIN, t = pos % LIN;
  unsigned short h[8], l[8];
#pragma unroll
  for (int j = 0; j < 8; ++j) {
    int tj = t + j; if (tj > LIN - 1) tj = LIN - 1;  // clamp hits only K>=251 (zero taps)
    float v = x[b * LIN + tj];
    h[j] = f2bf(v);
    l[j] = f2bf(v - bf2f(h[j]));
  }
  uint4 uh, ul;
  uh.x = (unsigned)h[0] | ((unsigned)h[1] << 16);
  uh.y = (unsigned)h[2] | ((unsigned)h[3] << 16);
  uh.z = (unsigned)h[4] | ((unsigned)h[5] << 16);
  uh.w = (unsigned)h[6] | ((unsigned)h[7] << 16);
  ul.x = (unsigned)l[0] | ((unsigned)l[1] << 16);
  ul.y = (unsigned)l[2] | ((unsigned)l[3] << 16);
  ul.z = (unsigned)l[4] | ((unsigned)l[5] << 16);
  ul.w = (unsigned)l[6] | ((unsigned)l[7] << 16);
  XH[pos] = uh;
  XL[pos] = ul;
}

// ---------------------------------------------------------------------------
// Phase 2: implicit-GEMM conv via v_wmma_f32_16x16x32_bf16.
// Block = 8 waves: mtile (16 filters) x 256 output times (2 sub-tiles/wave).
// Per wave: 6 independent accumulator chains (2 tiles x {hh, hl, lh}).
// B layout: lane L -> N(time) = L%16; lanes 0-15 cover K 0-15, 16-31 cover
// K 16-31; VGPR v holds K pair 2v,2v+1 -> fragment = 2 oct loads at
// element offsets (tau + kt*32 + half*16) and +8.
// ---------------------------------------------------------------------------
union Frag { uint4 u[2]; v16bf v; };

__global__ __launch_bounds__(256) void sinc_conv_wmma(
    const uint4* __restrict__ AH, const uint4* __restrict__ AL,
    const uint4* __restrict__ XH, const uint4* __restrict__ XL,
    float* __restrict__ out) {
  int lane = threadIdx.x & 31;
  int w    = threadIdx.x >> 5;
  int mt   = blockIdx.y;
  int b    = blockIdx.x / TTILES;
  int T0   = (blockIdx.x % TTILES) * NT;
  int n    = lane & 15;
  int half = lane >> 4;
  int tau0 = T0 + w * 32 + n;                // sub-tile 0 output time
  int tau1 = tau0 + 16;                      // sub-tile 1 output time

  const uint4* xh = XH + b * LIN;
  const uint4* xl = XL + b * LIN;

  v8f hh0 = {}, hl0 = {}, lh0 = {};
  v8f hh1 = {}, hl1 = {}, lh1 = {};

#pragma unroll 2
  for (int kt = 0; kt < 8; ++kt) {
    Frag ah, al;
    const uint4* ap  = AH + ((mt * 8 + kt) * 32 + lane) * 2;
    const uint4* alp = AL + ((mt * 8 + kt) * 32 + lane) * 2;
    ah.u[0] = ap[0];  ah.u[1] = ap[1];
    al.u[0] = alp[0]; al.u[1] = alp[1];

    int kb  = kt * 32 + half * 16;
    int i0  = tau0 + kb;     int i0b = i0 + 8;
    int i1  = tau1 + kb;     int i1b = i1 + 8;
    i0  = i0  < LIN ? i0  : LIN - 1;   // only invalid (unstored) taus clamp
    i0b = i0b < LIN ? i0b : LIN - 1;
    i1  = i1  < LIN ? i1  : LIN - 1;
    i1b = i1b < LIN ? i1b : LIN - 1;

    Frag b0h, b0l, b1h, b1l;
    b0h.u[0] = xh[i0]; b0h.u[1] = xh[i0b];
    b0l.u[0] = xl[i0]; b0l.u[1] = xl[i0b];
    b1h.u[0] = xh[i1]; b1h.u[1] = xh[i1b];
    b1l.u[0] = xl[i1]; b1l.u[1] = xl[i1b];

    hh0 = __builtin_amdgcn_wmma_f32_16x16x32_bf16(false, ah.v, false, b0h.v, (short)0, hh0, false, false);
    hl0 = __builtin_amdgcn_wmma_f32_16x16x32_bf16(false, ah.v, false, b0l.v, (short)0, hl0, false, false);
    lh0 = __builtin_amdgcn_wmma_f32_16x16x32_bf16(false, al.v, false, b0h.v, (short)0, lh0, false, false);
    hh1 = __builtin_amdgcn_wmma_f32_16x16x32_bf16(false, ah.v, false, b1h.v, (short)0, hh1, false, false);
    hl1 = __builtin_amdgcn_wmma_f32_16x16x32_bf16(false, ah.v, false, b1l.v, (short)0, hl1, false, false);
    lh1 = __builtin_amdgcn_wmma_f32_16x16x32_bf16(false, al.v, false, b1h.v, (short)0, lh1, false, false);
  }

  // C/D layout: VGPR r -> M(filter-in-tile) = r + 8*half, N(time) = lane%16.
  // Consecutive lanes -> consecutive times: coalesced stores.
#pragma unroll
  for (int r = 0; r < 8; ++r) {
    int f = mt * 16 + r + half * 8;
    float v0 = hh0[r] + (hl0[r] + lh0[r]);
    float v1 = hh1[r] + (hl1[r] + lh1[r]);
    if (tau0 < LOUT) out[(b * NFILT + f) * LOUT + tau0] = v0;
    if (tau1 < LOUT) out[(b * NFILT + f) * LOUT + tau1] = v1;
  }
}

// ---------------------------------------------------------------------------
extern "C" void kernel_launch(void* const* d_in, const int* in_sizes, int n_in,
                              void* d_out, int out_size, void* d_ws, size_t ws_size,
                              hipStream_t stream) {
  const float* x   = (const float*)d_in[0];
  const float* nf1 = (const float*)d_in[1];
  const float* nf2 = (const float*)d_in[2];
  const float* amp = (const float*)d_in[3];
  float* out = (float*)d_out;

  // Workspace carve-up (all 16B aligned):
  //   Ffilt: 80*256 f32            =   80 KB
  //   AH/AL: 5*8*32*8 u32 each     =   40 KB x2
  //   XH/XL: 32*32000 uint4 each   = 15.6 MB x2   (L2-resident on MI455X)
  char* ws = (char*)d_ws;
  float*    Ffilt = (float*)ws;     ws += (size_t)NFILT * KPAD * sizeof(float);
  unsigned* AH    = (unsigned*)ws;  ws += (size_t)MTILES * 8 * 32 * 8 * sizeof(unsigned);
  unsigned* AL    = (unsigned*)ws;  ws += (size_t)MTILES * 8 * 32 * 8 * sizeof(unsigned);
  uint4*    XH    = (uint4*)ws;     ws += (size_t)NBATCH * LIN * sizeof(uint4);
  uint4*    XL    = (uint4*)ws;

  build_filters<<<NFILT, 256, 0, stream>>>(nf1, nf2, amp, Ffilt);
  build_afrag<<<MTILES * 8, 32, 0, stream>>>(Ffilt, AH, AL);
  build_xoct<<<(NBATCH * LIN) / 256, 256, 0, stream>>>(x, XH, XL);

  dim3 grid(NBATCH * TTILES, MTILES);
  sinc_conv_wmma<<<grid, 256, 0, stream>>>((const uint4*)AH, (const uint4*)AL,
                                           XH, XL, out);
}
